// TemporalGCN_90555090469260
// MI455X (gfx1250) — compile-verified, matrix operand order checked by reading, not run on hardware
//
#include <hip/hip_runtime.h>
#include <cstdint>

// ---------------------------------------------------------------------------
// Types for CDNA5 WMMA (wave32): v_wmma_f32_16x16x32_bf16
// ---------------------------------------------------------------------------
typedef float          v8f   __attribute__((ext_vector_type(8)));
typedef __bf16         v16bf __attribute__((ext_vector_type(16)));
typedef unsigned short v8us  __attribute__((ext_vector_type(8)));
typedef unsigned short v16us __attribute__((ext_vector_type(16)));

__device__ __forceinline__ unsigned short f2bf(float f) {
  // round-to-nearest-even fp32 -> bf16 (bits)
  unsigned int u = __float_as_uint(f);
  u += 0x7FFFu + ((u >> 16) & 1u);
  return (unsigned short)(u >> 16);
}

__device__ __forceinline__ v16bf make_frag(v8us lo, v8us hi) {
  v16us t = __builtin_shufflevector(lo, hi, 0, 1, 2, 3, 4, 5, 6, 7,
                                    8, 9, 10, 11, 12, 13, 14, 15);
  return __builtin_bit_cast(v16bf, t);
}

// Load one 16x32 A/B fragment (CDNA5 ISA 7.12.2 16-bit layout) from a
// row-major [*,128] bf16-bits array given the per-lane row pointer.
__device__ __forceinline__ v16bf load_frag_row(const unsigned short* row, int kc, int kb) {
  v8us lo = *(const v8us*)&row[kc * 32 + kb];
  v8us hi = *(const v8us*)&row[kc * 32 + 16 + kb];
  return make_frag(lo, hi);
}

// ---------------------------------------------------------------------------
// WMMA GEMM:  C[M,Nc] (+)= A[M,128] * Bt[Nc,128]^T   (A fp32, Bt bf16-bits)
//   - M multiple of 16 (M = 50000 = 3125*16); K fixed 128 -> 4 WMMA / tile
//   - block 32x8 (8 waves): 16 rows per block, waves stride column tiles
//   - RELU_IN applies relu while staging A (fuses the GCN activation)
// ---------------------------------------------------------------------------
template <bool ACC, bool BIAS, bool RELU_IN>
__global__ __launch_bounds__(256) void gemm_k128(
    const float* __restrict__ A, const unsigned short* __restrict__ Bt,
    const float* __restrict__ bias, float* __restrict__ C, int Nc) {
  __shared__ alignas(16) unsigned short sA[16 * 136];

  const int tid = threadIdx.y * 32 + threadIdx.x;
  const int rowTile = blockIdx.x;

  // warm the weight stream for this wave's first column tile
  __builtin_prefetch(Bt + (size_t)(threadIdx.y * 16 + (threadIdx.x & 15)) * 128, 0, 1);

  // Stage 16x128 fp32 A tile (contiguous: lda==K==128) -> bf16 LDS
  const float* at = A + (size_t)rowTile * 16 * 128;
#pragma unroll
  for (int i = tid; i < 2048; i += 256) {
    float v = at[i];
    if (RELU_IN) v = fmaxf(v, 0.f);
    sA[(i >> 7) * 136 + (i & 127)] = f2bf(v);
  }
  __syncthreads();

  const int lane = threadIdx.x;
  const int r = lane & 15;         // row within tile (A) / col (B,C)
  const int kb = (lane >> 4) * 8;  // k sub-offset from lane[4]
  const int mb = (lane >> 4) * 8;  // C row sub-offset

  v16bf afrag[4];
  const unsigned short* ar = &sA[r * 136];
#pragma unroll
  for (int kc = 0; kc < 4; ++kc) afrag[kc] = load_frag_row(ar, kc, kb);

  const int nct = Nc >> 4;
  for (int ct = threadIdx.y; ct < nct; ct += 8) {
    const int col = ct * 16 + r;
    const unsigned short* br = Bt + (size_t)col * 128;
    v8f acc = {0.f, 0.f, 0.f, 0.f, 0.f, 0.f, 0.f, 0.f};
#pragma unroll
    for (int kc = 0; kc < 4; ++kc) {
      v16bf bfrag = load_frag_row(br, kc, kb);
      acc = __builtin_amdgcn_wmma_f32_16x16x32_bf16(
          false, afrag[kc], false, bfrag, (short)0, acc, false, false);
    }
#pragma unroll
    for (int v = 0; v < 8; ++v) {
      const size_t idx = (size_t)(rowTile * 16 + mb + v) * Nc + col;
      float val = acc[v];
      if (BIAS) val += bias[col];
      if (ACC) val += C[idx];
      C[idx] = val;
    }
  }
}

// ---------------------------------------------------------------------------
// Fused LSTM step (per 16-node row block):
//   G = relu(emb) @ Wih^T + h @ Whh^T        (all 32 col-tiles in LDS)
//   i,f,g,o gates -> c = sig(f)*c + sig(i)*tanh(g) ; h = sig(o)*tanh(c)
// G never touches global memory.
// ---------------------------------------------------------------------------
__global__ __launch_bounds__(256) void lstm_fused(
    const float* __restrict__ emb, const unsigned short* __restrict__ WihB,
    const unsigned short* __restrict__ WhhB, const float* __restrict__ bih,
    const float* __restrict__ bhh, float* __restrict__ h, float* __restrict__ c) {
  __shared__ alignas(16) unsigned short sE[16 * 136];
  __shared__ alignas(16) unsigned short sH[16 * 136];
  __shared__ float sG[16 * 520];  // 16 rows x 512 gates (+8 pad)

  const int tid = threadIdx.y * 32 + threadIdx.x;
  const int rowTile = blockIdx.x;

  __builtin_prefetch(WihB + (size_t)(threadIdx.y * 16 + (threadIdx.x & 15)) * 128, 0, 1);
  __builtin_prefetch(WhhB + (size_t)(threadIdx.y * 16 + (threadIdx.x & 15)) * 128, 0, 1);

  const float* et = emb + (size_t)rowTile * 16 * 128;
  const float* ht = h + (size_t)rowTile * 16 * 128;
#pragma unroll
  for (int i = tid; i < 2048; i += 256) {
    sE[(i >> 7) * 136 + (i & 127)] = f2bf(fmaxf(et[i], 0.f));  // relu fused
    sH[(i >> 7) * 136 + (i & 127)] = f2bf(ht[i]);
  }
  __syncthreads();

  const int lane = threadIdx.x;
  const int r = lane & 15;
  const int kb = (lane >> 4) * 8;
  const int mb = (lane >> 4) * 8;

  v16bf efrag[4], hfrag[4];
  {
    const unsigned short* er = &sE[r * 136];
    const unsigned short* hr = &sH[r * 136];
#pragma unroll
    for (int kc = 0; kc < 4; ++kc) {
      efrag[kc] = load_frag_row(er, kc, kb);
      hfrag[kc] = load_frag_row(hr, kc, kb);
    }
  }

#pragma unroll
  for (int ct = threadIdx.y; ct < 32; ct += 8) {  // Nc = 512 gate columns
    const int col = ct * 16 + r;
    const unsigned short* bi = WihB + (size_t)col * 128;
    const unsigned short* bh = WhhB + (size_t)col * 128;
    v8f acc = {0.f, 0.f, 0.f, 0.f, 0.f, 0.f, 0.f, 0.f};
#pragma unroll
    for (int kc = 0; kc < 4; ++kc) {
      v16bf bfrag = load_frag_row(bi, kc, kb);
      acc = __builtin_amdgcn_wmma_f32_16x16x32_bf16(
          false, efrag[kc], false, bfrag, (short)0, acc, false, false);
    }
#pragma unroll
    for (int kc = 0; kc < 4; ++kc) {
      v16bf bfrag = load_frag_row(bh, kc, kb);
      acc = __builtin_amdgcn_wmma_f32_16x16x32_bf16(
          false, hfrag[kc], false, bfrag, (short)0, acc, false, false);
    }
#pragma unroll
    for (int v = 0; v < 8; ++v) sG[(mb + v) * 520 + col] = acc[v];
  }
  __syncthreads();

  // gates: 16 rows x 128 features, 8 per thread
#pragma unroll
  for (int idx = tid; idx < 2048; idx += 256) {
    const int lr = idx >> 7, j = idx & 127;
    const float* g = &sG[lr * 520];
    const float gi = g[j] + bih[j] + bhh[j];
    const float gf = g[128 + j] + bih[128 + j] + bhh[128 + j];
    const float gg = g[256 + j] + bih[256 + j] + bhh[256 + j];
    const float go = g[384 + j] + bih[384 + j] + bhh[384 + j];
    const float si = 1.f / (1.f + expf(-gi));
    const float sf = 1.f / (1.f + expf(-gf));
    const float so = 1.f / (1.f + expf(-go));
    const size_t gidx = (size_t)rowTile * 2048 + idx;
    const float cc = sf * c[gidx] + si * tanhf(gg);
    c[gidx] = cc;
    h[gidx] = so * tanhf(cc);
  }
}

// ---------------------------------------------------------------------------
// Small helper kernels
// ---------------------------------------------------------------------------
__global__ void zero_f(float* p, int n) {
  int i = blockIdx.x * blockDim.x + threadIdx.x;
  if (i < n) p[i] = 0.f;
}
__global__ void fill1_f(float* p, int n) {
  int i = blockIdx.x * blockDim.x + threadIdx.x;
  if (i < n) p[i] = 1.f;  // self-loop contributes 1 to every degree
}
__global__ void copy_f(const float* __restrict__ s, float* __restrict__ d, int n) {
  int i = blockIdx.x * blockDim.x + threadIdx.x;
  if (i < n) d[i] = s[i];
}
// weights: fp32 [K][Nc] row-major -> bf16 [Nc][K] (B^T layout for GEMM)
__global__ void transpose_to_bf16(const float* __restrict__ in,
                                  unsigned short* __restrict__ out, int K, int Nc) {
  int idx = blockIdx.x * blockDim.x + threadIdx.x;
  if (idx >= K * Nc) return;
  int n = idx / K, k = idx - n * K;
  out[(size_t)n * K + k] = f2bf(in[(size_t)k * Nc + n]);
}
// Wih/Whh are [4H,H] row-major = already [Ncol][K]; just convert
__global__ void conv_to_bf16(const float* __restrict__ in,
                             unsigned short* __restrict__ out, int n) {
  int i = blockIdx.x * blockDim.x + threadIdx.x;
  if (i < n) out[i] = f2bf(in[i]);
}

// ---------------------------------------------------------------------------
// GCN aggregation: deg -> dinv, self-loop+bias init, edge scatter
// ---------------------------------------------------------------------------
__global__ void deg_count(const long long* __restrict__ dst, float* __restrict__ deg, int E) {
  int e = blockIdx.x * blockDim.x + threadIdx.x;
  if (e < E) atomicAdd(&deg[(int)dst[e]], 1.f);
}
__global__ void deg_finalize(float* d, int n) {
  int i = blockIdx.x * blockDim.x + threadIdx.x;
  if (i < n) d[i] = rsqrtf(d[i]);  // deg >= 1 always (self loop)
}
__global__ void gcn_self_bias(const float* __restrict__ xw, const float* __restrict__ dinv,
                              const float* __restrict__ b, float* __restrict__ agg, int NH) {
  int idx = blockIdx.x * blockDim.x + threadIdx.x;
  if (idx >= NH) return;
  int n = idx >> 7, j = idx & 127;
  float di = dinv[n];
  agg[idx] = xw[idx] * di * di + b[j];
}
__global__ void gcn_scatter(const long long* __restrict__ eit, const float* __restrict__ xw,
                            const float* __restrict__ dinv, float* __restrict__ agg, int E) {
  int e = blockIdx.x * blockDim.y + threadIdx.y;
  if (e >= E) return;
  int s = (int)eit[e];
  int d = (int)eit[(size_t)E + e];
  float coef = dinv[s] * dinv[d];
  int j = threadIdx.x;
  atomicAdd(&agg[(size_t)d * 128 + j], xw[(size_t)s * 128 + j] * coef);
}

// ---------------------------------------------------------------------------
// Host-side orchestration
// ---------------------------------------------------------------------------
extern "C" void kernel_launch(void* const* d_in, const int* in_sizes, int n_in,
                              void* d_out, int out_size, void* d_ws, size_t ws_size,
                              hipStream_t stream) {
  (void)n_in; (void)out_size; (void)ws_size;

  const float*     x_seq = (const float*)d_in[0];
  const long long* ei    = (const long long*)d_in[1];
  const float* W1   = (const float*)d_in[2];
  const float* b1   = (const float*)d_in[3];
  const float* W2   = (const float*)d_in[4];
  const float* b2   = (const float*)d_in[5];
  const float* Wih  = (const float*)d_in[6];
  const float* Whh  = (const float*)d_in[7];
  const float* bih  = (const float*)d_in[8];
  const float* bhh  = (const float*)d_in[9];
  const float* Wout = (const float*)d_in[10];
  const float* bout = (const float*)d_in[11];

  const int T = 8, N = 50000, F = 128, H = 128, O = 64;
  const int E = in_sizes[1] / (2 * T);
  const int NH = N * H;

  // workspace carve-out (256B aligned slices)
  char* w = (char*)d_ws;
  auto carve = [&](size_t bytes) {
    void* p = (void*)w;
    w += (bytes + 255) & ~(size_t)255;
    return p;
  };
  float* dinv = (float*)carve((size_t)N * 4);
  float* xw   = (float*)carve((size_t)NH * 4);
  float* act1 = (float*)carve((size_t)NH * 4);  // pre-relu GCN1 output
  float* emb  = (float*)carve((size_t)NH * 4);  // pre-relu GCN2 output
  float* h    = (float*)carve((size_t)NH * 4);
  float* c    = (float*)carve((size_t)NH * 4);
  unsigned short* W1t   = (unsigned short*)carve((size_t)F * H * 2);
  unsigned short* W2t   = (unsigned short*)carve((size_t)H * H * 2);
  unsigned short* Woutt = (unsigned short*)carve((size_t)H * O * 2);
  unsigned short* WihB  = (unsigned short*)carve((size_t)4 * H * H * 2);
  unsigned short* WhhB  = (unsigned short*)carve((size_t)4 * H * H * 2);

  const dim3 gblk(32, 8);
  const int gTiles = N / 16;  // 3125
  const int TB = 256;
  auto nb = [](int n, int b) { return (n + b - 1) / b; };

  // prologue: zero state, prep bf16 weights (B^T layout)
  zero_f<<<nb(NH, TB), TB, 0, stream>>>(h, NH);
  zero_f<<<nb(NH, TB), TB, 0, stream>>>(c, NH);
  transpose_to_bf16<<<nb(F * H, TB), TB, 0, stream>>>(W1, W1t, F, H);
  transpose_to_bf16<<<nb(H * H, TB), TB, 0, stream>>>(W2, W2t, H, H);
  transpose_to_bf16<<<nb(H * O, TB), TB, 0, stream>>>(Wout, Woutt, H, O);
  conv_to_bf16<<<nb(4 * H * H, TB), TB, 0, stream>>>(Wih, WihB, 4 * H * H);
  conv_to_bf16<<<nb(4 * H * H, TB), TB, 0, stream>>>(Whh, WhhB, 4 * H * H);

  const dim3 sblk(128, 4);  // feature x edges-per-block
  for (int t = 0; t < T; ++t) {
    const long long* eit = ei + (size_t)t * 2 * E;
    const float* xt = x_seq + (size_t)t * N * F;

    // symmetric-normalization coefficients for this timestep
    fill1_f<<<nb(N, TB), TB, 0, stream>>>(dinv, N);
    deg_count<<<nb(E, TB), TB, 0, stream>>>(eit + E, dinv, E);
    deg_finalize<<<nb(N, TB), TB, 0, stream>>>(dinv, N);

    // GCN layer 1 (act1 holds pre-relu; relu fused into the next GEMM's A stage)
    gemm_k128<false, false, false><<<gTiles, gblk, 0, stream>>>(xt, W1t, nullptr, xw, H);
    gcn_self_bias<<<nb(NH, TB), TB, 0, stream>>>(xw, dinv, b1, act1, NH);
    gcn_scatter<<<nb(E, 4), sblk, 0, stream>>>(eit, xw, dinv, act1, E);

    // GCN layer 2 (reads relu(act1) via RELU_IN staging; emb holds pre-relu)
    gemm_k128<false, false, true><<<gTiles, gblk, 0, stream>>>(act1, W2t, nullptr, xw, H);
    gcn_self_bias<<<nb(NH, TB), TB, 0, stream>>>(xw, dinv, b2, emb, NH);
    gcn_scatter<<<nb(E, 4), sblk, 0, stream>>>(eit, xw, dinv, emb, E);

    // fused LSTM step: G kept entirely in LDS, gates applied in-kernel
    lstm_fused<<<gTiles, gblk, 0, stream>>>(emb, WihB, WhhB, bih, bhh, h, c);
  }

  // epilogue: out = h @ Wout + bout ; then hT ; then cT
  float* out = (float*)d_out;
  gemm_k128<false, true, false><<<gTiles, gblk, 0, stream>>>(h, Woutt, bout, out, O);
  copy_f<<<nb(NH, TB), TB, 0, stream>>>(h, out + (size_t)N * O, NH);
  copy_f<<<nb(NH, TB), TB, 0, stream>>>(c, out + (size_t)N * O + NH, NH);
}
